// CLSTMSegmenter_79379585565447
// MI455X (gfx1250) — matrix operand order repaired and checked by reading, not compiled, not run on hardware
//
#include <hip/hip_runtime.h>

// ---------------------------------------------------------------------------
// CDNA5 (gfx1250, wave32) ConvLSTM segmenter.
// Gate conv as implicit GEMM with v_wmma_f32_16x16x32_bf16.
// Staging via global_load_async_to_lds_b128 (ASYNCcnt path).
// ---------------------------------------------------------------------------

typedef __attribute__((ext_vector_type(4)))  unsigned int  u32x4;
typedef __attribute__((ext_vector_type(16))) __bf16        v16bf;
typedef __attribute__((ext_vector_type(8)))  float         v8f;

union Frag32 { u32x4 u[2]; v16bf v; };   // 32 bytes = one wmma bf16 A/B fragment per lane

#define NB    8
#define NTS   12
#define CIN   64
#define HIDC  128
#define HH    64
#define WW    64
#define NCLS  5
#define PIX   4096              // H*W
#define LDSPXB 400              // bytes per pixel record (192ch*2B padded -> b128 conflict-free)
#define HALOW  66               // 64 px row + 2 halo columns
#define LDSRECS (3 * HALOW)     // 198 pixel records

__device__ __forceinline__ unsigned short f2bf(float f) {
  unsigned int u = __float_as_uint(f);
  u += 0x7FFFu + ((u >> 16) & 1u);      // round-to-nearest-even
  return (unsigned short)(u >> 16);
}
__device__ __forceinline__ float bf2f(unsigned short h) {
  return __uint_as_float(((unsigned int)h) << 16);
}
__device__ __forceinline__ float sigm(float x) { return 1.0f / (1.0f + __expf(-x)); }

// Async DMA: global (16B) -> LDS, tracked by ASYNCcnt. Per-lane addresses.
__device__ __forceinline__ void async_g2l_b128(unsigned int ldsOff, const void* g) {
  unsigned long long ga = (unsigned long long)(size_t)g;
  asm volatile("global_load_async_to_lds_b128 %0, %1, off"
               :: "v"(ldsOff), "v"(ga)
               : "memory");
}
__device__ __forceinline__ void wait_asynccnt0() {
  asm volatile("s_wait_asynccnt 0x0" ::: "memory");
}

// ---------------------------------------------------------------------------
// 1) Transpose x: (B,T,C,H,W) fp32 -> pixel-major (B*T, H*W, C) bf16.  LDS-tiled.
// ---------------------------------------------------------------------------
__global__ __launch_bounds__(256) void convert_x(const float* __restrict__ x,
                                                 unsigned short* __restrict__ xbf) {
  __shared__ unsigned short tile[64][65];
  const int blk = blockIdx.x;           // BT*64 blocks
  const int bt = blk >> 6;
  const int p0 = (blk & 63) << 6;       // 64-pixel slab
  for (int i = threadIdx.x; i < 4096; i += 256) {
    const int c = i >> 6, p = i & 63;   // coalesced read along pixels
    tile[c][p] = f2bf(x[((size_t)bt * CIN + c) * PIX + p0 + p]);
  }
  __syncthreads();
  for (int i = threadIdx.x; i < 4096; i += 256) {
    const int p = i >> 6, c = i & 63;   // coalesced write along channels
    xbf[((size_t)bt * PIX + p0 + p) * CIN + c] = tile[c][p];
  }
}

// ---------------------------------------------------------------------------
// 2) Pack w_lstm (512,192,3,3) fp32 into bf16 A-fragments in the exact
//    v_wmma 16-bit A 16x32 lane layout (ISA 7.12.2):
//    lane<16: M=lane,   K in {0..7} (V0-3) and {16..23} (V4-7)
//    lane>=16:M=lane-16,K in {8..15}        and {24..31}
//    fragment id = ((kh*3+kw)*6 + kc)*32 + (gate*8 + wave)
// ---------------------------------------------------------------------------
__global__ __launch_bounds__(256) void pack_wlstm(const float* __restrict__ w,
                                                  unsigned short* __restrict__ wpk) {
  const int pid = blockIdx.x * 256 + threadIdx.x;        // one packed ushort each
  if (pid >= 54 * 32 * 32 * 16) return;
  const int frag   = pid >> 9;
  const int rem    = pid & 511;
  const int lane   = rem >> 4;
  const int within = rem & 15;
  const int v  = within >> 1;
  const int hh = within & 1;
  const int kstep = frag >> 5;
  const int mtile = frag & 31;
  const int g = mtile >> 3, wv = mtile & 7;
  const int oc = g * 128 + wv * 16 + (lane & 15);
  const int kh = kstep / 18;
  const int kw = (kstep / 6) % 3;
  const int kc = kstep % 6;
  const int klocal = ((v < 4) ? 2 * v : 2 * v + 8) + ((lane < 16) ? 0 : 8) + hh;
  const int ic = kc * 32 + klocal;      // 0..191 within concat(x,h)
  wpk[pid] = f2bf(w[((size_t)oc * 192 + ic) * 9 + kh * 3 + kw]);
}

// ---------------------------------------------------------------------------
// 3) Zero initial h0 (bf16) and c (fp32): laid out adjacently in workspace.
// ---------------------------------------------------------------------------
__global__ __launch_bounds__(256) void zero_ws(unsigned int* __restrict__ p, int nwords) {
  for (int i = blockIdx.x * 256 + threadIdx.x; i < nwords; i += gridDim.x * 256) p[i] = 0u;
}

// ---------------------------------------------------------------------------
// 4) One ConvLSTM step. Grid: 512 workgroups = B * H; 256 threads (8 waves).
//    Each workgroup: one full 64-pixel row, all 512 gate channels.
//    Wave w owns hid rows [16w,16w+16) for all four gates -> pointwise LSTM
//    update happens directly on the accumulator fragments.
//    Per wave: 54 k-steps x (4 gates x 4 N-tiles) = 864 WMMAs, 128 acc VGPRs.
// ---------------------------------------------------------------------------
__global__ __launch_bounds__(256) void clstm_step(
    const unsigned short* __restrict__ xbf,
    const unsigned short* __restrict__ hprev,
    const unsigned short* __restrict__ wpk,
    const float* __restrict__ blstm,
    float* __restrict__ cbuf,
    unsigned short* __restrict__ hnext,
    float* __restrict__ hfp,            // fp32 h (written only on last step)
    int t) {
  __shared__ __align__(16) unsigned char smem[LDSRECS * LDSPXB];   // 79,200 B

  const int wg  = blockIdx.x;           // 0..511
  const int b   = wg >> 6;
  const int y   = wg & 63;
  const int tid = threadIdx.x;
  const int bt  = b * NTS + t;

  // ---- Stage combined (x_t ++ h_prev) halo tile: 3 rows x 66 px x 192 bf16 ch.
  // Pixel-major records, channel-contiguous, 400B stride.
  // In-bounds chunks: async DMA global->LDS (ASYNCcnt); padding: ds_store zeros.
  for (int task = tid; task < LDSRECS * 24; task += 256) {
    const int rec = task / 24;
    const int ch  = (task % 24) << 3;         // 8 bf16 channels per 16B chunk
    const int row = rec / HALOW;
    const int xl  = rec % HALOW;
    const int gy = y + row - 1;
    const int gx = xl - 1;
    const unsigned int dst = (unsigned int)(size_t)(smem + rec * LDSPXB + ch * 2);
    if (gy >= 0 && gy < HH && gx >= 0 && gx < WW) {
      const void* src = (ch < CIN)
          ? (const void*)(xbf + ((size_t)bt * PIX + gy * WW + gx) * CIN + ch)
          : (const void*)(hprev + ((size_t)b * PIX + gy * WW + gx) * HIDC + (ch - CIN));
      async_g2l_b128(dst, src);
    } else {
      const u32x4 z = {0u, 0u, 0u, 0u};       // SAME padding
      *(u32x4*)(smem + rec * LDSPXB + ch * 2) = z;
    }
  }
  wait_asynccnt0();
  __syncthreads();

  const int wave  = tid >> 5;                 // wave32
  const int lane  = tid & 31;
  const int npix  = lane & 15;                // N index within 16-wide tile
  const int lhalf = lane >> 4;                // 0: lanes 0-15, 1: lanes 16-31

  v8f acc[4][4];
  #pragma unroll
  for (int g = 0; g < 4; ++g) {
    #pragma unroll
    for (int nt = 0; nt < 4; ++nt) acc[g][nt] = {};
  }

  for (int kh = 0; kh < 3; ++kh) {
    for (int kw = 0; kw < 3; ++kw) {
      #pragma unroll
      for (int kc = 0; kc < 6; ++kc) {
        // B fragments (pixels x channels) from LDS; lanes 0-15 K=0..7,16..23, 16-31 K=8..15,24..31
        Frag32 bfr[4];
        {
          const int choff = kc * 64 + lhalf * 16;           // byte offset in record
          const unsigned char* p0 = smem + (kh * HALOW + npix + kw) * LDSPXB + choff;
          #pragma unroll
          for (int nt = 0; nt < 4; ++nt) {
            const unsigned char* p = p0 + nt * 16 * LDSPXB;
            bfr[nt].u[0] = *(const u32x4*)(p);
            bfr[nt].u[1] = *(const u32x4*)(p + 32);
          }
        }
        const int kstep = (kh * 3 + kw) * 6 + kc;
        #pragma unroll
        for (int g = 0; g < 4; ++g) {
          Frag32 af;                                        // prepacked, L2-resident
          const unsigned short* ap =
              wpk + (((size_t)(kstep * 32 + g * 8 + wave) * 32 + lane) << 4);
          af.u[0] = *(const u32x4*)(ap);
          af.u[1] = *(const u32x4*)(ap + 8);
          #pragma unroll
          for (int nt = 0; nt < 4; ++nt)
            acc[g][nt] = __builtin_amdgcn_wmma_f32_16x16x32_bf16(
                false, af.v, false, bfr[nt].v, (short)0, acc[g][nt], false, false);
        }
      }
    }
  }

  // ---- LSTM pointwise update directly on C/D fragments.
  // C/D layout: lane<16 -> M=e, lane>=16 -> M=e+8 ; N = lane%16.
  const int hidBase = wave * 16;
  #pragma unroll
  for (int nt = 0; nt < 4; ++nt) {
    const int gx = nt * 16 + npix;
    const size_t base = ((size_t)b * PIX + y * WW + gx) * HIDC;
    #pragma unroll
    for (int e = 0; e < 8; ++e) {
      const int hid = hidBase + e + lhalf * 8;
      const float gi = acc[0][nt][e] + blstm[hid];
      const float gf = acc[1][nt][e] + blstm[128 + hid];
      const float go = acc[2][nt][e] + blstm[256 + hid];
      const float gg = acc[3][nt][e] + blstm[384 + hid];
      const float iv = sigm(gi), fv = sigm(gf), ov = sigm(go);
      const float gv = tanhf(gg);
      const float cn = fv * cbuf[base + hid] + iv * gv;     // c stays fp32
      cbuf[base + hid] = cn;
      const float hn = ov * tanhf(cn);
      hnext[base + hid] = f2bf(hn);                         // bf16 for next GEMM
      if (hfp) hfp[base + hid] = hn;                        // fp32 for the head
    }
  }
}

// ---------------------------------------------------------------------------
// 5) Head: 3x3 conv 128->5 on fp32 h, then channelwise log_softmax.
//    ~3.8e8 FLOPs total -> plain VALU kernel is fine.
// ---------------------------------------------------------------------------
__global__ __launch_bounds__(256) void final_conv(
    const float* __restrict__ hfp, const float* __restrict__ wconv,
    const float* __restrict__ bconv, float* __restrict__ out) {
  const int pid = blockIdx.x * 256 + threadIdx.x;
  if (pid >= NB * PIX) return;
  const int b = pid >> 12;
  const int p = pid & 4095;
  const int y = p >> 6, x = p & 63;
  float acc[NCLS];
  #pragma unroll
  for (int k = 0; k < NCLS; ++k) acc[k] = bconv[k];
  for (int kh = 0; kh < 3; ++kh) {
    const int gy = y + kh - 1;
    if (gy < 0 || gy >= HH) continue;
    for (int kw = 0; kw < 3; ++kw) {
      const int gx = x + kw - 1;
      if (gx < 0 || gx >= WW) continue;
      const float* hp = hfp + ((size_t)b * PIX + gy * WW + gx) * HIDC;
      const float* wp = wconv + kh * 3 + kw;
      for (int c = 0; c < HIDC; ++c) {
        const float hv = hp[c];
        #pragma unroll
        for (int k = 0; k < NCLS; ++k)
          acc[k] = fmaf(hv, wp[((size_t)k * HIDC + c) * 9], acc[k]);
      }
    }
  }
  float m = acc[0];
  #pragma unroll
  for (int k = 1; k < NCLS; ++k) m = fmaxf(m, acc[k]);
  float s = 0.f;
  #pragma unroll
  for (int k = 0; k < NCLS; ++k) s += __expf(acc[k] - m);
  const float lse = __logf(s);
  #pragma unroll
  for (int k = 0; k < NCLS; ++k)
    out[((size_t)b * NCLS + k) * PIX + p] = acc[k] - m - lse;
}

// ---------------------------------------------------------------------------
// Workspace layout (bytes):
//   xbf  @ 0         : 50,331,648  (B*T*4096*64 bf16, pixel-major)
//   h0   @ 50331648  :  8,388,608  (bf16 pixel-major)   -- adjacent to c
//   c    @ 58720256  : 16,777,216  (fp32 pixel-major)      for one zero pass
//   h1   @ 75497472  :  8,388,608
//   wpk  @ 83886080  :  1,769,472  (packed bf16 A-fragments)
//   hfp  @ 85655552  : 16,777,216  (fp32 final h)
// Total ~97.7 MB.
// ---------------------------------------------------------------------------
extern "C" void kernel_launch(void* const* d_in, const int* in_sizes, int n_in,
                              void* d_out, int out_size, void* d_ws, size_t ws_size,
                              hipStream_t stream) {
  const float* x     = (const float*)d_in[0];
  const float* wlstm = (const float*)d_in[1];
  const float* blstm = (const float*)d_in[2];
  const float* wconv = (const float*)d_in[3];
  const float* bconv = (const float*)d_in[4];
  float* out = (float*)d_out;
  char* ws = (char*)d_ws;

  unsigned short* xbf = (unsigned short*)(ws);
  unsigned short* h0  = (unsigned short*)(ws + 50331648);
  float*          cb  = (float*)(ws + 58720256);
  unsigned short* h1  = (unsigned short*)(ws + 75497472);
  unsigned short* wpk = (unsigned short*)(ws + 83886080);
  float*          hfp = (float*)(ws + 85655552);

  convert_x <<<NB * NTS * (PIX / 64), 256, 0, stream>>>(x, xbf);       // 6144 blocks
  pack_wlstm<<<(54 * 32 * 32 * 16) / 256, 256, 0, stream>>>(wlstm, wpk);
  zero_ws   <<<2048, 256, 0, stream>>>((unsigned int*)h0, 25165824 / 4); // h0 + c

  for (int t = 0; t < NTS; ++t) {
    const unsigned short* hp = (t & 1) ? h1 : h0;
    unsigned short* hn       = (t & 1) ? h0 : h1;
    clstm_step<<<NB * HH, 256, 0, stream>>>(
        xbf, hp, wpk, blstm, cb, hn, (t == NTS - 1) ? hfp : nullptr, t);
  }
  final_conv<<<NB * PIX / 256, 256, 0, stream>>>(hfp, wconv, bconv, out);
}
